// SparseAutoencoder_34780645163129
// MI455X (gfx1250) — matrix operand order repaired and checked by reading, not compiled, not run on hardware
//
#include <hip/hip_runtime.h>

// Problem constants (match reference)
#define N_ROWS   32768
#define D_MODEL  1280
#define D_HIDDEN 4096
#define K_TOP    32
#define EPS_LN   1e-5f

// WMMA vector types (per CDNA5_HIP.md probe-confirmed signatures)
typedef __attribute__((ext_vector_type(16))) __bf16 v16bf;
typedef __attribute__((ext_vector_type(8)))  float  v8f;

union Frag32B { v16bf v; uint4 q[2]; };

// f32 -> bf16 bits, round-to-nearest-even
__device__ __forceinline__ unsigned short f2bf(float f) {
  unsigned u = __float_as_uint(f);
  unsigned r = (u + 0x7FFFu + ((u >> 16) & 1u)) >> 16;
  return (unsigned short)r;
}

// ------------------------------------------------------------------
// Kernel 1: w_enc [1280][4096] f32 (row-major) -> column-major bf16
//           wet [4096][1280]  (wet[n][k] = bf16(w_enc[k][n]))
// Tiled LDS transpose for coalescing on both sides.
// ------------------------------------------------------------------
__global__ __launch_bounds__(256)
void wenc_transpose_bf16(const float* __restrict__ w,
                         unsigned short* __restrict__ wt) {
  __shared__ float tile[32][33];
  const int n0 = blockIdx.x * 32;
  const int k0 = blockIdx.y * 32;
  const int tx = threadIdx.x;          // 0..31
  const int ty = threadIdx.y;          // 0..7
#pragma unroll
  for (int i = 0; i < 32; i += 8)
    tile[ty + i][tx] = w[(size_t)(k0 + ty + i) * D_HIDDEN + (n0 + tx)];
  __syncthreads();
#pragma unroll
  for (int i = 0; i < 32; i += 8)
    wt[(size_t)(n0 + ty + i) * D_MODEL + (k0 + tx)] = f2bf(tile[tx][ty + i]);
}

// ------------------------------------------------------------------
// Kernel 2: fused LN -> encode GEMM (WMMA bf16) -> top-K -> decode.
// One 256-thread workgroup (8 wave32) per 16 rows.
//
// Dynamic LDS layout (bytes):
//   [0,       262144)  pre_acts  : f32 [16][4096]   (phase 0 reuses as x f32)
//   [262144,  303360)  A strip   : bf16 [16][1288]  (padded stride -> no bank conflicts)
//   [303360,  303424)  mu  [16]
//   [303424,  303488)  std [16]
//   [303488,  305536)  vals[16][32] f32
//   [305536,  307584)  idx [16][32] i32
// ------------------------------------------------------------------
#define A_STRIDE 1288  // elements; 2576 B row stride, 16B-aligned, bank-skewed

__global__ __launch_bounds__(256)
void sae_fused(const float* __restrict__ x,
               const unsigned short* __restrict__ wet,  // [4096][1280] bf16 bits
               const float* __restrict__ w_dec,         // [4096][1280] f32
               const float* __restrict__ b_enc,         // [4096]
               const float* __restrict__ b_pre,         // [1280]
               float* __restrict__ out) {               // [32768][1280]
  extern __shared__ char smem[];
  float*          preacts = (float*)smem;                             // 16 x 4096
  unsigned short* Astrip  = (unsigned short*)(smem + 262144);         // 16 x 1288
  float*          muL     = (float*)(smem + 303360);
  float*          stdL    = (float*)(smem + 303424);
  float*          valsL   = (float*)(smem + 303488);                  // 16 x 32
  int*            idxL    = (int*)  (smem + 305536);                  // 16 x 32

  const int tid  = threadIdx.x;
  const int lane = tid & 31;
  const int wv   = tid >> 5;          // wave id 0..7
  const int r0   = blockIdx.x * 16;   // first row of this block

  // ---------- Phase 0: stage 16 rows of x (f32) into LDS ----------
  float* xbuf = preacts;  // reuse pre_acts region (80 KB of it)
  for (int m = 0; m < 16; ++m)
    for (int c = tid; c < D_MODEL; c += 256)
      xbuf[m * D_MODEL + c] = x[(size_t)(r0 + m) * D_MODEL + c];
  __syncthreads();

  // ---------- Phase 1: per-row mean / std (Bessel, ddof=1) ----------
#pragma unroll
  for (int rr = 0; rr < 2; ++rr) {
    const int m = 2 * wv + rr;
    float s = 0.f, ss = 0.f;
    for (int c = lane; c < D_MODEL; c += 32) {
      float v = xbuf[m * D_MODEL + c];
      s += v;
      ss += v * v;
    }
#pragma unroll
    for (int off = 16; off > 0; off >>= 1) {
      s  += __shfl_xor(s, off, 32);
      ss += __shfl_xor(ss, off, 32);
    }
    const float mu  = s / (float)D_MODEL;
    float var = (ss - (float)D_MODEL * mu * mu) / (float)(D_MODEL - 1);
    var = fmaxf(var, 0.f);
    if (lane == 0) { muL[m] = mu; stdL[m] = sqrtf(var); }
  }
  __syncthreads();

  // ---------- Phase 2: bf16 A strip = (x - mu)/(std+eps) - b_pre ----------
  for (int m = 0; m < 16; ++m) {
    const float mu  = muL[m];
    const float inv = 1.f / (stdL[m] + EPS_LN);
    for (int c = tid; c < D_MODEL; c += 256) {
      float v = (xbuf[m * D_MODEL + c] - mu) * inv - b_pre[c];
      Astrip[m * A_STRIDE + c] = f2bf(v);
    }
  }
  __syncthreads();

  // ---------- Phase 3: encode GEMM, 16 rows x 4096 cols, K = 1280 ----------
  // Each wave owns 512 columns: 32 tiles of 16 cols, 40 WMMA k-steps each.
  {
    const int g  = lane >> 4;       // lane half: selects K sub-range
    const int nl = lane & 15;       // local column / local row id
    const int colBase = wv * 512;
    for (int nt = 0; nt < 32; ++nt) {
      const int n0 = colBase + nt * 16;
      v8f acc = {};
      const unsigned short* bcol = wet + (size_t)(n0 + nl) * D_MODEL;
      const unsigned short* arow = Astrip + nl * A_STRIDE;
      for (int k0 = 0; k0 < D_MODEL; k0 += 32) {
        Frag32B a, b;
        // A (16x32 bf16): lane half g -> K = {g*8..g*8+7} U {16+g*8..16+g*8+7}
        a.q[0] = *(const uint4*)(arow + k0 + g * 8);
        a.q[1] = *(const uint4*)(arow + k0 + 16 + g * 8);
        // B (32x16 bf16, column-major in memory): lane half g -> K = {g*8..g*8+7} U {16+g*8..16+g*8+7}
        b.q[0] = *(const uint4*)(bcol + k0 + g * 8);
        b.q[1] = *(const uint4*)(bcol + k0 + 16 + g * 8);
        acc = __builtin_amdgcn_wmma_f32_16x16x32_bf16(
            false, a.v, false, b.v, (short)0, acc, false, false);
      }
      // C layout: lane -> col n0+nl, VGPR r -> row r + 8*g
      const float be = b_enc[n0 + nl];
#pragma unroll
      for (int r = 0; r < 8; ++r)
        preacts[(r + 8 * g) * D_HIDDEN + (n0 + nl)] = acc[r] + be;
    }
  }
  __syncthreads();

  // ---------- Phase 4: exact top-32 per row (iterative wave argmax) ----------
  {
    volatile float* pa = (volatile float*)preacts;
#pragma unroll
    for (int rr = 0; rr < 2; ++rr) {
      const int m = 2 * wv + rr;
      volatile float* row = pa + m * D_HIDDEN;
      for (int j = 0; j < K_TOP; ++j) {
        float bv = -3.4e38f;
        int   bi = 0x7FFFFFFF;
        for (int c = lane; c < D_HIDDEN; c += 32) {
          float v = row[c];
          if (v > bv || (v == bv && c < bi)) { bv = v; bi = c; }
        }
#pragma unroll
        for (int off = 16; off > 0; off >>= 1) {
          float ov = __shfl_xor(bv, off, 32);
          int   oi = __shfl_xor(bi, off, 32);
          if (ov > bv || (ov == bv && oi < bi)) { bv = ov; bi = oi; }
        }
        if (lane == 0) {
          valsL[m * K_TOP + j] = fmaxf(bv, 0.f);   // ReLU on selected values
          idxL [m * K_TOP + j] = bi;
          row[bi] = -3.4e38f;                      // remove winner
        }
        // DS ops are in-order within a wave; volatile keeps compiler order.
      }
    }
  }
  __syncthreads();

  // ---------- Phase 5: sparse decode + un-normalize ----------
#pragma unroll
  for (int rr = 0; rr < 2; ++rr) {
    const int m  = 2 * wv + rr;
    const float mu = muL[m];
    const float sd = stdL[m];
    for (int d = lane; d < D_MODEL; d += 32) {
      float acc = b_pre[d];
#pragma unroll 4
      for (int j = 0; j < K_TOP; ++j) {
        const float v = valsL[m * K_TOP + j];
        if (v != 0.f) {
          const int id = idxL[m * K_TOP + j];
          acc += v * w_dec[(size_t)id * D_MODEL + d];
        }
      }
      out[(size_t)(r0 + m) * D_MODEL + d] = acc * sd + mu;
    }
  }
}

// ------------------------------------------------------------------
extern "C" void kernel_launch(void* const* d_in, const int* in_sizes, int n_in,
                              void* d_out, int out_size, void* d_ws, size_t ws_size,
                              hipStream_t stream) {
  const float* x     = (const float*)d_in[0];  // [32768,1280]
  const float* w_enc = (const float*)d_in[1];  // [1280,4096]
  const float* w_dec = (const float*)d_in[2];  // [4096,1280]
  const float* b_enc = (const float*)d_in[3];  // [4096]
  const float* b_pre = (const float*)d_in[4];  // [1280]
  (void)in_sizes; (void)n_in; (void)out_size; (void)ws_size;

  // Workspace: bf16 column-major encoder weights, 4096*1280*2 = 10.5 MB
  unsigned short* wet = (unsigned short*)d_ws;

  dim3 tb(32, 8);
  dim3 tg(D_HIDDEN / 32, D_MODEL / 32);
  wenc_transpose_bf16<<<tg, tb, 0, stream>>>(w_enc, wet);

  const size_t lds_bytes = 307584;  // see layout comment above (<320 KB WGP LDS)
  sae_fused<<<N_ROWS / 16, 256, lds_bytes, stream>>>(
      x, wet, w_dec, b_enc, b_pre, (float*)d_out);
}
